// Net_76235669504605
// MI455X (gfx1250) — matrix operand order repaired and checked by reading, not compiled, not run on hardware
//
#include <hip/hip_runtime.h>

typedef _Float16 v16h __attribute__((ext_vector_type(16)));
typedef _Float16 v8h  __attribute__((ext_vector_type(8)));
typedef _Float16 v4h  __attribute__((ext_vector_type(4)));
typedef float    v8f  __attribute__((ext_vector_type(8)));
typedef float    v4f  __attribute__((ext_vector_type(4)));

#define BETA 0.95f
#define THR  1.0f

#define B_   16384
#define NI_  256
#define NH_  512
#define NO_  128

// ---------------------------------------------------------------------------
// Phase 0: one-time split of an f32 array into hi/lo f16 arrays.
// hi = f16(v), lo = f16(v - f32(hi))  ->  hi+lo carries ~22 mantissa bits.
// ---------------------------------------------------------------------------
__global__ __launch_bounds__(256) void split_f32_f16(const float* __restrict__ src,
                                                     _Float16* __restrict__ hi,
                                                     _Float16* __restrict__ lo,
                                                     int n)
{
    const int i = (blockIdx.x * 256 + threadIdx.x) * 4;
    if (i < n) {
        v4f v = *(const v4f*)(src + i);
        v4h h, l;
        #pragma unroll
        for (int e = 0; e < 4; ++e) {
            _Float16 hh = (_Float16)v[e];
            h[e] = hh;
            l[e] = (_Float16)(v[e] - (float)hh);
        }
        *(v4h*)(hi + i) = h;
        *(v4h*)(lo + i) = l;
    }
}

// ---------------------------------------------------------------------------
// Phase 1: cur1 = x @ w1^T + b1   (split-f16 WMMA, 3 terms: hh + hl + lh)
// One wave computes a 16x64 output tile (4 N-tiles); K=256 in 8 k-tiles.
// Inner loop is pure f16 loads + WMMA (conversions were done in phase 0).
// ---------------------------------------------------------------------------
__global__ __launch_bounds__(256) void snn_gemm1(const _Float16* __restrict__ xhi,
                                                 const _Float16* __restrict__ xlo,
                                                 const _Float16* __restrict__ w1hi,
                                                 const _Float16* __restrict__ w1lo,
                                                 const float* __restrict__ b1,
                                                 float* __restrict__ cur1)
{
    const int lane  = threadIdx.x & 31;
    const int wave  = threadIdx.x >> 5;
    const int gw    = blockIdx.x * 8 + wave;   // 8192 wave-tiles total
    const int mtile = gw >> 3;                 // 0..1023 (16-row tiles)
    const int ntg   = gw & 7;                  // 0..7    (64-col groups)

    const int  l16  = lane & 15;
    const bool hh   = lane >= 16;
    const int  mrow = (mtile << 4) + l16;

    v8f acc[4] = {};

    #pragma unroll
    for (int kt = 0; kt < NI_ / 32; ++kt) {
        // A fragment (16x32 f16): lane-half K base 0/16; elems 0..15 -> K=ka+e
        // (contiguous). Pre-split arrays are plain row-major.
        const int ka = kt * 32 + (hh ? 16 : 0);
        const _Float16* axh = xhi + (size_t)mrow * NI_ + ka;
        const _Float16* axl = xlo + (size_t)mrow * NI_ + ka;
        v8h h0 = *(const v8h*)(axh);
        v8h h1 = *(const v8h*)(axh + 8);
        v8h l0 = *(const v8h*)(axl);
        v8h l1 = *(const v8h*)(axl + 8);
        v16h Ahi, Alo;
        #pragma unroll
        for (int e = 0; e < 8; ++e) {
            Ahi[e] = h0[e]; Ahi[e + 8] = h1[e];
            Alo[e] = l0[e]; Alo[e + 8] = l1[e];
        }
        // B fragments (32x16): lane = N, lane-half K base 0/16, K linear.
        const int kb = kt * 32 + (hh ? 16 : 0);
        #pragma unroll
        for (int nt = 0; nt < 4; ++nt) {
            const int ncol = ntg * 64 + nt * 16 + l16;
            v16h Bh = *(const v16h*)(w1hi + (size_t)ncol * NI_ + kb);
            v16h Bl = *(const v16h*)(w1lo + (size_t)ncol * NI_ + kb);
            acc[nt] = __builtin_amdgcn_wmma_f32_16x16x32_f16(false, Ahi, false, Bh, (short)0, acc[nt], false, false);
            acc[nt] = __builtin_amdgcn_wmma_f32_16x16x32_f16(false, Ahi, false, Bl, (short)0, acc[nt], false, false);
            acc[nt] = __builtin_amdgcn_wmma_f32_16x16x32_f16(false, Alo, false, Bh, (short)0, acc[nt], false, false);
        }
    }

    // C/D layout: VGPR r -> row r (lanes 0-15) / r+8 (lanes 16-31), N = lane&15
    const int rbase = (mtile << 4) + (hh ? 8 : 0);
    #pragma unroll
    for (int nt = 0; nt < 4; ++nt) {
        const int  ncol = ntg * 64 + nt * 16 + l16;
        const float bias = b1[ncol];
        #pragma unroll
        for (int r = 0; r < 8; ++r)
            cur1[(size_t)(rbase + r) * NH_ + ncol] = acc[nt][r] + bias;
    }
}

// ---------------------------------------------------------------------------
// Phase 2: fused 32-step LIF loop.
// Block = 256 threads (8 waves) owns 32 batch rows.
//   LDS: w2 hi/lo f16 [128][512] (256 KB) + spk1 f16 [32][512] (32 KB)
//   Regs: per-thread cur1/m1 slice (row = tid>>3, 64 cols), m2 in acc layout.
//   Wave w computes N-tile w for both 16-row M-tiles; B fragments are loaded
//   once per k-tile and feed both M-tile accumulator chains (interleaved).
// ---------------------------------------------------------------------------
__global__ __launch_bounds__(256) void snn_steps(const float* __restrict__ cur1,
                                                 const _Float16* __restrict__ w2hi,
                                                 const _Float16* __restrict__ w2lo,
                                                 const float* __restrict__ b2,
                                                 const int* __restrict__ nsteps_p,
                                                 float* __restrict__ out)
{
    extern __shared__ char smem[];
    _Float16* lds_w2hi = (_Float16*)smem;            // [128][512]
    _Float16* lds_w2lo = lds_w2hi + NO_ * NH_;       // [128][512]
    _Float16* lds_spk  = lds_w2lo + NO_ * NH_;       // [32][512]

    const int tid     = threadIdx.x;
    const int lane    = tid & 31;
    const int wave    = tid >> 5;
    const int rowbase = blockIdx.x * 32;
    const int T       = *nsteps_p;                   // device scalar (32)

    // ---- preload pre-split w2 into LDS (pure f16 copy) ----
    for (int i = tid * 8; i < NO_ * NH_; i += 256 * 8) {
        *(v8h*)(lds_w2hi + i) = *(const v8h*)(w2hi + i);
        *(v8h*)(lds_w2lo + i) = *(const v8h*)(w2lo + i);
    }

    // ---- per-thread cur1 / m1 state in registers ----
    const int myrow  = tid >> 3;            // 0..31
    const int mycol0 = (tid & 7) * 64;      // 8 threads per row, 64 cols each
    float c1[64];
    float m1[64];
    {
        const float* src = cur1 + (size_t)(rowbase + myrow) * NH_ + mycol0;
        #pragma unroll
        for (int j = 0; j < 16; ++j) {
            v4f v = *(const v4f*)(src + 4 * j);
            #pragma unroll
            for (int e = 0; e < 4; ++e) { c1[4 * j + e] = v[e]; m1[4 * j + e] = 0.0f; }
        }
    }

    // ---- wave-private GEMM2 constants + m2 state (accumulator layout) ----
    const int  l16   = lane & 15;
    const bool hh    = lane >= 16;
    const int  ncol  = wave * 16 + l16;     // output neuron for this lane
    const float biasn = b2[ncol];
    float m2[2][8] = {};                    // [mtile][vgpr r]

    __syncthreads();

    for (int t = 0; t < T; ++t) {
        // ---- LIF layer 1 (subtract reset), spikes -> LDS as exact f16 ----
        _Float16 sp[64];
        #pragma unroll
        for (int j = 0; j < 64; ++j) {
            float reset = (m1[j] > THR) ? 1.0f : 0.0f;
            m1[j] = BETA * m1[j] + c1[j] - reset * THR;
            sp[j] = (m1[j] > THR) ? (_Float16)1.0f : (_Float16)0.0f;
        }
        #pragma unroll
        for (int c = 0; c < 8; ++c)
            *(v8h*)(lds_spk + myrow * NH_ + mycol0 + 8 * c) = *(const v8h*)(sp + 8 * c);

        __syncthreads();

        // ---- cur2 = spk1 @ w2^T : B fragments shared by both M-tiles ----
        v8f acc[2] = {};
        {
            const _Float16* a0p = lds_spk + (size_t)(l16) * NH_;        // M-tile 0
            const _Float16* a1p = lds_spk + (size_t)(16 + l16) * NH_;   // M-tile 1
            const _Float16* bhp = lds_w2hi + (size_t)ncol * NH_;
            const _Float16* blp = lds_w2lo + (size_t)ncol * NH_;
            #pragma unroll
            for (int kt = 0; kt < NH_ / 32; ++kt) {
                const int ka = kt * 32 + (hh ? 16 : 0);
                v8h x00 = *(const v8h*)(a0p + ka);
                v8h x01 = *(const v8h*)(a0p + ka + 8);
                v8h x10 = *(const v8h*)(a1p + ka);
                v8h x11 = *(const v8h*)(a1p + ka + 8);
                v16h A0, A1;
                #pragma unroll
                for (int e = 0; e < 8; ++e) {
                    A0[e] = x00[e]; A0[e + 8] = x01[e];
                    A1[e] = x10[e]; A1[e + 8] = x11[e];
                }
                const int kb = kt * 32 + (hh ? 16 : 0);
                v16h Bh = *(const v16h*)(bhp + kb);
                v16h Bl = *(const v16h*)(blp + kb);
                acc[0] = __builtin_amdgcn_wmma_f32_16x16x32_f16(false, A0, false, Bh, (short)0, acc[0], false, false);
                acc[1] = __builtin_amdgcn_wmma_f32_16x16x32_f16(false, A1, false, Bh, (short)0, acc[1], false, false);
                acc[0] = __builtin_amdgcn_wmma_f32_16x16x32_f16(false, A0, false, Bl, (short)0, acc[0], false, false);
                acc[1] = __builtin_amdgcn_wmma_f32_16x16x32_f16(false, A1, false, Bl, (short)0, acc[1], false, false);
            }
        }

        // ---- LIF layer 2 + stream spk2/mem2 to HBM ----
        #pragma unroll
        for (int mt = 0; mt < 2; ++mt) {
            const int rbase = rowbase + mt * 16 + (hh ? 8 : 0);
            #pragma unroll
            for (int r = 0; r < 8; ++r) {
                float cur2  = acc[mt][r] + biasn;
                float mm    = m2[mt][r];
                float reset = (mm > THR) ? 1.0f : 0.0f;
                mm = BETA * mm + cur2 - reset * THR;
                float spike = (mm > THR) ? 1.0f : 0.0f;
                m2[mt][r] = mm;
                size_t o = (size_t)t * B_ * NO_ + (size_t)(rbase + r) * NO_ + ncol;
                out[o] = spike;                              // spk2_rec
                out[(size_t)T * B_ * NO_ + o] = mm;          // mem2_rec
            }
        }
        __syncthreads();   // protect lds_spk before next step overwrites it
    }
}

// ---------------------------------------------------------------------------
extern "C" void kernel_launch(void* const* d_in, const int* in_sizes, int n_in,
                              void* d_out, int out_size, void* d_ws, size_t ws_size,
                              hipStream_t stream)
{
    (void)in_sizes; (void)n_in; (void)out_size; (void)ws_size;
    const float* x   = (const float*)d_in[0];
    const float* w1  = (const float*)d_in[1];
    const float* b1  = (const float*)d_in[2];
    const float* w2  = (const float*)d_in[3];
    const float* b2  = (const float*)d_in[4];
    const int*   nst = (const int*)d_in[5];
    float* out = (float*)d_out;

    // Workspace layout (bytes):              offset        size
    char* ws = (char*)d_ws;
    float*    cur1 = (float*)(ws);                        // 0          33,554,432
    _Float16* xhi  = (_Float16*)(ws + 33554432);          // 32 MB       8,388,608
    _Float16* xlo  = (_Float16*)(ws + 41943040);          //             8,388,608
    _Float16* w1hi = (_Float16*)(ws + 50331648);          //               262,144
    _Float16* w1lo = (_Float16*)(ws + 50593792);          //               262,144
    _Float16* w2hi = (_Float16*)(ws + 50855936);          //               131,072
    _Float16* w2lo = (_Float16*)(ws + 50987008);          //               131,072
    // total ~48.8 MB

    // Phase 0: one-time hi/lo splits (elementwise, 4 floats per thread)
    split_f32_f16<<<(B_ * NI_) / 1024, 256, 0, stream>>>(x,  xhi,  xlo,  B_ * NI_);
    split_f32_f16<<<(NH_ * NI_) / 1024, 256, 0, stream>>>(w1, w1hi, w1lo, NH_ * NI_);
    split_f32_f16<<<(NO_ * NH_) / 1024, 256, 0, stream>>>(w2, w2hi, w2lo, NO_ * NH_);

    // Phase 1: 8192 wave-tiles (16x64 each) / 8 waves per block = 1024 blocks
    snn_gemm1<<<1024, 256, 0, stream>>>(xhi, xlo, w1hi, w1lo, b1, cur1);

    // Phase 2: 512 blocks of 32 rows; 288 KB dynamic LDS (<= 320 KB/WGP)
    hipFuncSetAttribute((const void*)snn_steps,
                        hipFuncAttributeMaxDynamicSharedMemorySize, 294912);
    snn_steps<<<512, 256, 294912, stream>>>(cur1, w2hi, w2lo, b2, nst, out);
}
